// Attention_979252544074
// MI455X (gfx1250) — matrix operand order repaired and checked by reading, not compiled, not run on hardware
//
#include <hip/hip_runtime.h>

typedef __attribute__((ext_vector_type(16))) _Float16 v16h;
typedef __attribute__((ext_vector_type(8)))  float    v8f;
typedef __attribute__((ext_vector_type(4)))  unsigned int v4u;
typedef __attribute__((ext_vector_type(8)))  int      v8i;
typedef __attribute__((ext_vector_type(4)))  int      v4i;

#if defined(__has_builtin)
#if __has_builtin(__builtin_amdgcn_tensor_load_to_lds)
#define HAVE_TDM 1
#endif
#endif
#ifndef HAVE_TDM
#define HAVE_TDM 0
#endif

#define NTOK   49
#define NPAD   64
#define DIM    256
#define HEADS  8
#define DH     32
#define NMEM   4
#define NKV    53

// ---------------- LDS layout (bytes) ----------------
#define XN_STRIDE   264
#define AOUT_STRIDE 264
#define QH_STRIDE   40
#define KH_STRIDE   40
#define VT_STRIDE   72
#define PH_STRIDE   72

#define XN_OFF   0                         // f16 [64][264]  = 33792 B
#define AOUT_OFF 33792                     // f16 [64][264]  = 33792 B
#define QH_OFF   67584                     // f16 [64][40]   =  5120 B
#define KH_OFF   72704                     // f16 [64][40]   =  5120 B
#define VT_OFF   77824                     // f16 [32][72]   =  4608 B
#define PF_OFF   82432                     // f32 [64][64]   = 16384 B
#define PH_OFF   98816                     // f16 [64][72]   =  9216 B
#define LDS_BYTES 108032

// x staging area (fp32 window, 50176 B) overlays AOUT..PF (65024 B, dead at
// that point in the schedule): offsets [33792, 83968)
#define XF_OFF   AOUT_OFF

// ---------------- workspace layout (bytes) ----------------
#define WS_WQKVT 0                         // f16 [768][256] (transposed, q pre-scaled)
#define WS_WOUTT 393216                    // f16 [256][256] (transposed)
#define WS_MEMH  524288                    // f16 [2][8][4][32]
#define WS_BIAS  528384                    // f32 [8][49][64]

// ---------------- WMMA fragment loaders ----------------
// A fragment (16x32 f16): lane holds row M=lane&15; lane-group (lane>>4) adds
// k-offset 8; per lane two contiguous 8-half chunks at k+0..7 and k+16..23.
__device__ __forceinline__ v16h load_A_frag(const _Float16* base, int stride,
                                            int row0, int k0) {
  const int lane = threadIdx.x & 31;
  const _Float16* p = base + (row0 + (lane & 15)) * stride + k0 + ((lane >> 4) << 3);
  v16h a;
#pragma unroll
  for (int e = 0; e < 16; ++e) {
    const int j = e >> 1;
    a[e] = p[((j & 4) ? 16 : 0) + 2 * (j & 3) + (e & 1)];
  }
  return a;
}

// B fragment (32x16 f16) from a k-contiguous layout [col][k] (stride per col):
// lane holds column N=lane&15; lane-group adds k-offset 8; chunks k+0..7, k+16..23.
__device__ __forceinline__ v16h load_B_frag(const _Float16* base, int stride,
                                            int col0, int k0) {
  const int lane = threadIdx.x & 31;
  const _Float16* p = base + (col0 + (lane & 15)) * stride + k0 + ((lane >> 4) << 3);
  v16h b;
#pragma unroll
  for (int e = 0; e < 16; ++e) b[e] = p[(e < 8) ? e : e + 8];
  return b;
}

// ---------------- prep: transpose/convert weights, expand bias ----------------
__global__ __launch_bounds__(256) void prep_kernel(
    const float* __restrict__ wqkv, const float* __restrict__ wout,
    const float* __restrict__ memkv, const float* __restrict__ btab,
    _Float16* __restrict__ wqkvt, _Float16* __restrict__ woutt,
    _Float16* __restrict__ memh, float* __restrict__ bias) {
  const int NW1 = 768 * 256, NW2 = 256 * 256, NM = 2 * HEADS * NMEM * DH;
  const int NB = HEADS * NTOK * 64;
  const int total = NW1 + NW2 + NM + NB;
  for (int idx = blockIdx.x * 256 + threadIdx.x; idx < total;
       idx += gridDim.x * 256) {
    if (idx < NW1) {
      int col = idx >> 8, k = idx & 255;
      float v = wqkv[k * 768 + col];
      if (col < 256) v *= 0.17677669529663687f;   // q * dh^-0.5 folded into Wq
      wqkvt[idx] = (_Float16)v;
    } else if (idx < NW1 + NW2) {
      int i = idx - NW1;
      int col = i >> 8, k = i & 255;
      woutt[i] = (_Float16)wout[k * 256 + col];
    } else if (idx < NW1 + NW2 + NM) {
      int i = idx - NW1 - NW2;
      memh[i] = (_Float16)memkv[i];
    } else {
      int i = idx - NW1 - NW2 - NM;
      int h = i / (NTOK * 64), r = i % (NTOK * 64);
      int ti = r >> 6, j = r & 63;
      float v = 0.f;
      if (j >= NMEM && j < NKV) {
        int jj = j - NMEM;
        int rel = (ti / 7 - jj / 7 + 6) * 13 + (ti % 7 - jj % 7 + 6);
        v = btab[rel * HEADS + h];
      }
      bias[i] = v;
    }
  }
}

// ---------------- fused window-attention kernel ----------------
__global__ __launch_bounds__(256) void attn_win_kernel(
    const float* __restrict__ x, const float* __restrict__ gamma,
    const _Float16* __restrict__ wqkvt, const _Float16* __restrict__ memh,
    const float* __restrict__ bias, const _Float16* __restrict__ woutt,
    float* __restrict__ out) {
  extern __shared__ char smem[];
  _Float16* XN   = (_Float16*)(smem + XN_OFF);
  _Float16* AOUT = (_Float16*)(smem + AOUT_OFF);
  _Float16* QH   = (_Float16*)(smem + QH_OFF);
  _Float16* KH   = (_Float16*)(smem + KH_OFF);
  _Float16* VT   = (_Float16*)(smem + VT_OFF);
  float*    PF   = (float*)   (smem + PF_OFF);
  _Float16* PH   = (_Float16*)(smem + PH_OFF);
  float*    XF   = (float*)   (smem + XF_OFF);   // staged fp32 x window

  const int w    = blockIdx.x;
  const int tid  = threadIdx.x;
  const int wv   = tid >> 5;
  const int lane = tid & 31;

  const float* xw = x + (size_t)w * NTOK * DIM;

  // ---- stage x window into LDS via Tensor Data Mover (async DMA) ----
#if HAVE_TDM
  if (wv == 0) {
    unsigned long long ga = (unsigned long long)xw;
    // D# group0: count=1 | lds_addr | global_addr[56:0] | type=2
    v4u g0 = { 1u, (unsigned)XF_OFF, (unsigned)(ga & 0xFFFFFFFFu),
               (unsigned)(((ga >> 32) & 0x01FFFFFFu) | (2u << 30)) };
    // D# group1: data_size=4B; tensor_dim0=tile_dim0=49*256=12544 (0x3100);
    // tensor_dim1=tile_dim1=1; dim0_stride=12544; no pad/iterate/multicast.
    v8i g1 = { (int)(2u << 16),            // data_size=2 (4 bytes)
               (int)(0x3100u << 16),       // tensor_dim0[15:0]
               (int)(1u << 16),            // tensor_dim0[31:16]=0, tensor_dim1=1
               (int)(0x3100u << 16),       // tensor_dim1 hi=0, tile_dim0=12544
               1,                          // tile_dim1=1, tile_dim2=0
               12544,                      // tensor_dim0_stride[31:0]
               0, 0 };                     // strides hi = 0
    v4i z4 = { 0, 0, 0, 0 };
#if __clang_major__ >= 23
    v8i z8 = { 0, 0, 0, 0, 0, 0, 0, 0 };
    __builtin_amdgcn_tensor_load_to_lds(g0, g1, z4, z4, z8, 0);
#else
    __builtin_amdgcn_tensor_load_to_lds(g0, g1, z4, z4, 0);
#endif
  }
#else
  for (int i = tid; i < NTOK * DIM; i += 256) XF[i] = xw[i];
#endif

  // zero XN padding rows while the DMA is in flight (disjoint LDS region)
  for (int i = tid; i < (NPAD - NTOK) * XN_STRIDE; i += 256)
    XN[NTOK * XN_STRIDE + i] = (_Float16)0.f;

#if HAVE_TDM
  if (wv == 0) {
#if __has_builtin(__builtin_amdgcn_s_wait_tensorcnt)
    __builtin_amdgcn_s_wait_tensorcnt(0);
#else
    asm volatile("s_wait_tensorcnt 0x0" ::: "memory");
#endif
  }
#endif
  __syncthreads();

  // ---- LayerNorm (from LDS-staged x) -> XN (f16) ----
  for (int t = wv; t < NTOK; t += 8) {
    float vals[8], s = 0.f, sq = 0.f;
#pragma unroll
    for (int u = 0; u < 8; ++u) {
      float v = XF[t * DIM + lane + 32 * u];
      vals[u] = v; s += v; sq += v * v;
    }
#pragma unroll
    for (int off = 16; off; off >>= 1) {
      s  += __shfl_xor(s, off, 32);
      sq += __shfl_xor(sq, off, 32);
    }
    float mu = s * (1.f / 256.f);
    float rs = rsqrtf(sq * (1.f / 256.f) - mu * mu + 1e-5f);
#pragma unroll
    for (int u = 0; u < 8; ++u) {
      int c = lane + 32 * u;
      XN[t * XN_STRIDE + c] = (_Float16)((vals[u] - mu) * rs * gamma[c]);
    }
  }
  __syncthreads();   // XF region is now dead; safe to reuse as QH/KH/VT/AOUT/PF

  // zero padded rows of the per-head buffers (token rows >= 49 never written)
  for (int i = tid; i < 64 * QH_STRIDE; i += 256) {
    QH[i] = (_Float16)0.f;
    KH[i] = (_Float16)0.f;
  }
  for (int i = tid; i < 32 * VT_STRIDE; i += 256) VT[i] = (_Float16)0.f;
  __syncthreads();

  // ---- per-head attention ----
  for (int h = 0; h < HEADS; ++h) {
    // memory kv rows (KH rows 0..3, VT cols 0..3)
    if (tid < 128) {
      int mr = tid >> 5, mc = tid & 31;
      KH[mr * KH_STRIDE + mc] =
          memh[((0 * HEADS + h) * NMEM + mr) * DH + mc];
    } else {
      int t2 = tid - 128, mr = t2 >> 5, mc = t2 & 31;
      VT[mc * VT_STRIDE + mr] =
          memh[((1 * HEADS + h) * NMEM + mr) * DH + mc];
    }

    // Q/K/V projection for this head: 24 output tiles, 3 per wave
#pragma unroll
    for (int it = 0; it < 3; ++it) {
      int tix = wv * 3 + it;
      int mat = tix >> 3, t8 = tix & 7, mt = t8 >> 1, nt = t8 & 1;
      int col0 = mat * DIM + h * DH + nt * 16;
      v8f acc = {};
#pragma unroll
      for (int kk = 0; kk < 8; ++kk) {
        v16h a = load_A_frag(XN, XN_STRIDE, mt * 16, kk * 32);
        v16h b = load_B_frag(wqkvt, DIM, col0, kk * 32);
        acc = __builtin_amdgcn_wmma_f32_16x16x32_f16(false, a, false, b,
                                                     (short)0, acc, false, false);
      }
      int nb = nt * 16 + (lane & 15);
      int mb = (lane >> 4) << 3;
#pragma unroll
      for (int r = 0; r < 8; ++r) {
        int m = mt * 16 + mb + r;
        _Float16 hv = (_Float16)acc[r];
        if (mat == 0) {
          QH[m * QH_STRIDE + nb] = hv;      // padded Q rows feed zeroed P rows
        } else if (m < NTOK) {
          if (mat == 1) KH[(NMEM + m) * KH_STRIDE + nb] = hv;
          else          VT[nb * VT_STRIDE + (NMEM + m)] = hv;
        }
      }
    }
    __syncthreads();

    // sim = Q @ K^T (+bias): 16 tiles, 2 per wave, single K=32 step
#pragma unroll
    for (int it = 0; it < 2; ++it) {
      int tix = wv * 2 + it, mt = tix >> 2, nt = tix & 3;
      v16h a = load_A_frag(QH, QH_STRIDE, mt * 16, 0);
      v16h b = load_B_frag(KH, KH_STRIDE, nt * 16, 0);
      v8f acc = {};
      acc = __builtin_amdgcn_wmma_f32_16x16x32_f16(false, a, false, b,
                                                   (short)0, acc, false, false);
      int nb = nt * 16 + (lane & 15);
      int mb = (lane >> 4) << 3;
#pragma unroll
      for (int r = 0; r < 8; ++r) {
        int i = mt * 16 + mb + r;
        float v = acc[r];
        if (i < NTOK) v += bias[(h * NTOK + i) * 64 + nb];
        PF[i * 64 + nb] = v;
      }
    }
    __syncthreads();

    // softmax over 53 keys (rows >= 49 forced to zero)
    for (int t = wv; t < NPAD; t += 8) {
      if (t >= NTOK) {
        PH[t * PH_STRIDE + lane] = (_Float16)0.f;
        PH[t * PH_STRIDE + lane + 32] = (_Float16)0.f;
      } else {
        float v0 = (lane < NKV) ? PF[t * 64 + lane] : -1e30f;
        float v1 = -1e30f;
        if (lane + 32 < NKV) v1 = PF[t * 64 + lane + 32];
        float m = fmaxf(v0, v1);
#pragma unroll
        for (int off = 16; off; off >>= 1) m = fmaxf(m, __shfl_xor(m, off, 32));
        float e0 = __expf(v0 - m), e1 = __expf(v1 - m);
        float s = e0 + e1;
#pragma unroll
        for (int off = 16; off; off >>= 1) s += __shfl_xor(s, off, 32);
        float inv = 1.f / s;
        PH[t * PH_STRIDE + lane]      = (_Float16)(e0 * inv);
        PH[t * PH_STRIDE + lane + 32] = (_Float16)(e1 * inv);
      }
    }
    __syncthreads();

    // out_h = P @ V : 8 tiles, 1 per wave, K=64 (2 steps)
    {
      int mt = wv >> 1, nt = wv & 1;
      v8f acc = {};
#pragma unroll
      for (int ks = 0; ks < 2; ++ks) {
        v16h a = load_A_frag(PH, PH_STRIDE, mt * 16, ks * 32);
        v16h b = load_B_frag(VT, VT_STRIDE, nt * 16, ks * 32);
        acc = __builtin_amdgcn_wmma_f32_16x16x32_f16(false, a, false, b,
                                                     (short)0, acc, false, false);
      }
      int nb = nt * 16 + (lane & 15);
      int mb = (lane >> 4) << 3;
#pragma unroll
      for (int r = 0; r < 8; ++r)
        AOUT[(mt * 16 + mb + r) * AOUT_STRIDE + h * DH + nb] = (_Float16)acc[r];
    }
    __syncthreads();
  }

  // ---- output projection: AOUT[64x256] @ Wout[256x256] -> fp32 out ----
  float* outw = out + (size_t)w * NTOK * DIM;
#pragma unroll
  for (int it = 0; it < 8; ++it) {
    int tix = wv * 8 + it, mt = tix >> 4, nt = tix & 15;
    v8f acc = {};
#pragma unroll
    for (int kk = 0; kk < 8; ++kk) {
      v16h a = load_A_frag(AOUT, AOUT_STRIDE, mt * 16, kk * 32);
      v16h b = load_B_frag(woutt, DIM, nt * 16, kk * 32);
      acc = __builtin_amdgcn_wmma_f32_16x16x32_f16(false, a, false, b,
                                                   (short)0, acc, false, false);
    }
    int nb = nt * 16 + (lane & 15);
    int mb = (lane >> 4) << 3;
#pragma unroll
    for (int r = 0; r < 8; ++r) {
      int m = mt * 16 + mb + r;
      if (m < NTOK) outw[m * DIM + nb] = acc[r];
    }
  }
}

extern "C" void kernel_launch(void* const* d_in, const int* in_sizes, int n_in,
                              void* d_out, int out_size, void* d_ws, size_t ws_size,
                              hipStream_t stream) {
  const float* x     = (const float*)d_in[0];
  const float* gamma = (const float*)d_in[1];
  const float* wqkv  = (const float*)d_in[2];
  const float* memkv = (const float*)d_in[3];
  const float* btab  = (const float*)d_in[4];
  const float* wout  = (const float*)d_in[5];
  float* out = (float*)d_out;

  char* ws = (char*)d_ws;
  _Float16* wqkvt = (_Float16*)(ws + WS_WQKVT);
  _Float16* woutt = (_Float16*)(ws + WS_WOUTT);
  _Float16* memh  = (_Float16*)(ws + WS_MEMH);
  float*    bias  = (float*)   (ws + WS_BIAS);

  (void)in_sizes; (void)n_in; (void)out_size; (void)ws_size;

  hipFuncSetAttribute(reinterpret_cast<const void*>(attn_win_kernel),
                      hipFuncAttributeMaxDynamicSharedMemorySize, LDS_BYTES);

  prep_kernel<<<1130, 256, 0, stream>>>(wqkv, wout, memkv, btab,
                                        wqkvt, woutt, memh, bias);
  attn_win_kernel<<<4096, 256, LDS_BYTES, stream>>>(x, gamma, wqkvt, memh,
                                                    bias, woutt, out);
}
